// AttentionHead_54176717472257
// MI455X (gfx1250) — matrix operand order repaired and checked by reading, not compiled, not run on hardware
//
#include <hip/hip_runtime.h>
#include <stdint.h>

// ---------------------------------------------------------------------------
// Causal single-head attention for MI455X (gfx1250, wave32, WMMA).
//   x: [8, 2048, 1024] f32;  Wq/Wk/Wv: [1024, 64] f32;  out: [8, 2048, 64] f32
// Pipeline:
//   1) wconv : W* -> wt f16 [192][1024]   (transposed, B-frag friendly)
//   2) qkv   : x tile -> LDS via global_load_async_to_lds_b128 (ASYNCcnt),
//              f16 WMMA GEMM; q,k row-major f16, v transposed f16
//   3) flash : online-softmax flash attention, f16 WMMA, f32 accumulate
// ---------------------------------------------------------------------------

typedef __attribute__((ext_vector_type(16))) _Float16 v16h;
typedef __attribute__((ext_vector_type(8)))  _Float16 v8h;
typedef __attribute__((ext_vector_type(8)))  float    v8f;

#define D_MODEL 1024
#define HEAD    64
#define TSEQ    2048
#define BATCH   8

static __device__ __forceinline__ v8f wmma_f16(v16h a, v16h b, v8f c) {
    return __builtin_amdgcn_wmma_f32_16x16x32_f16(
        /*neg_a=*/false, a, /*neg_b=*/false, b,
        /*c_mod=*/(short)0, c, /*reuse_a=*/false, /*reuse_b=*/false);
}

// Async copy 16B from global to LDS (GV mode, ASYNCcnt-tracked, no VGPR data).
static __device__ __forceinline__ void async_copy_b128(uint32_t lds_off,
                                                       uint64_t gaddr) {
    asm volatile("global_load_async_to_lds_b128 %0, %1, off"
                 :: "v"(lds_off), "v"(gaddr) : "memory");
}

// -------------------------------------------------------------------- wconv
__global__ __launch_bounds__(256)
void attn_wconv_kernel(const float* __restrict__ Wq, const float* __restrict__ Wk,
                       const float* __restrict__ Wv, _Float16* __restrict__ wt) {
    int idx = blockIdx.x * blockDim.x + threadIdx.x;   // 0 .. 192*1024-1
    int col = idx >> 10;                               // 0..191
    int kk  = idx & 1023;
    const float* W = (col < 64) ? Wq : (col < 128) ? Wk : Wv;
    int h = col & 63;
    wt[(size_t)col * D_MODEL + kk] = (_Float16)W[(size_t)kk * HEAD + h];
}

// ---------------------------------------------------------------------- qkv
// block = 96 threads (3 waves). Block owns a 16-row tile of x, staged as f32
// in LDS by the async-copy engine; wave 0 -> q, wave 1 -> k, wave 2 -> v.
__global__ __launch_bounds__(96)
void attn_qkv_kernel(const float* __restrict__ x, const _Float16* __restrict__ wt,
                     _Float16* __restrict__ qf, _Float16* __restrict__ kf,
                     _Float16* __restrict__ vt) {
    __shared__ float xs[16 * D_MODEL];                 // 64 KB, f32 tile
    const int tid     = threadIdx.x;
    const int rowbase = blockIdx.x * 16;

    // 16 rows x 1024 f32 is one contiguous 64 KB span of x: pure async DMA,
    // no VGPR landing zone, tracked on ASYNCcnt.
    {
        const float* src = x + (size_t)rowbase * D_MODEL;
        const uint32_t lds_base = (uint32_t)(uintptr_t)&xs[0];
        for (int e4 = tid; e4 < (16 * D_MODEL) / 4; e4 += 96) {
            async_copy_b128(lds_base + (uint32_t)e4 * 16u,
                            (uint64_t)(uintptr_t)(src + (size_t)e4 * 4));
        }
        asm volatile("s_wait_asynccnt 0x0" ::: "memory");
    }
    __syncthreads();

    const int wid  = tid >> 5;          // 0:q 1:k 2:v
    const int lane = tid & 31;
    const int half = lane >> 4;
    const int n16  = lane & 15;

    v8f acc[4] = {};
    const _Float16* wcol = wt + (size_t)wid * 64 * D_MODEL;

    for (int k0 = 0; k0 < D_MODEL; k0 += 32) {
        // A fragment (16x32 f16): two contiguous 8-value chunks, f32 in LDS,
        // converted to f16 in registers (overlaps with WMMA issue).
        const int abase = n16 * D_MODEL + k0 + half * 16;
        v8f c1 = *(const v8f*)&xs[abase];
        v8f c2 = *(const v8f*)&xs[abase + 8];
        v16h a;
        #pragma unroll
        for (int i = 0; i < 8; ++i) {
            a[i]     = (_Float16)c1[i];
            a[i + 8] = (_Float16)c2[i];
        }

        #pragma unroll
        for (int nt = 0; nt < 4; ++nt) {
            // B fragment (32x16): lane = column, 16 contiguous K values.
            const _Float16* bp =
                wcol + (size_t)(nt * 16 + n16) * D_MODEL + k0 + half * 16;
            v16h b = *(const v16h*)bp;
            acc[nt] = wmma_f16(a, b, acc[nt]);
        }
    }

    if (wid < 2) {                      // q, k: row-major f16 [B*T][64]
        _Float16* dst = (wid == 0) ? qf : kf;
        #pragma unroll
        for (int nt = 0; nt < 4; ++nt)
            #pragma unroll
            for (int i = 0; i < 8; ++i)
                dst[(size_t)(rowbase + i + 8 * half) * HEAD + nt * 16 + n16] =
                    (_Float16)acc[nt][i];
    } else {                            // v: transposed f16 [b][64][2048]
        const int b  = rowbase >> 11;
        const int t0 = rowbase & (TSEQ - 1);
        #pragma unroll
        for (int nt = 0; nt < 4; ++nt) {
            v8h h8;
            #pragma unroll
            for (int i = 0; i < 8; ++i) h8[i] = (_Float16)acc[nt][i];
            _Float16* dst = vt + ((size_t)b * HEAD + nt * 16 + n16) * TSEQ
                               + t0 + 8 * half;
            *(v8h*)dst = h8;            // 8 consecutive t values -> b128 store
        }
    }
}

// -------------------------------------------------------------------- flash
// block = 128 threads (4 waves); each wave owns one 16-query tile and streams
// 32-key tiles with online softmax. All matmuls are v_wmma_f32_16x16x32_f16.
__global__ __launch_bounds__(128)
void attn_flash_kernel(const _Float16* __restrict__ qf, const _Float16* __restrict__ kf,
                       const _Float16* __restrict__ vt, float* __restrict__ out) {
    __shared__ _Float16 Pbuf[4][16 * 32];              // 1 KB per wave
    const int tid  = threadIdx.x;
    const int wid  = tid >> 5;
    const int lane = tid & 31;
    const int half = lane >> 4;
    const int n16  = lane & 15;

    const int qt    = blockIdx.x * 4 + wid;            // 0..1023
    const int b     = qt >> 7;
    const int qbase = (qt & 127) << 4;

    const _Float16* qrow = qf + (size_t)b * TSEQ * HEAD;
    const _Float16* krow = kf + (size_t)b * TSEQ * HEAD;
    const _Float16* vtb  = vt + (size_t)b * HEAD * TSEQ;

    // Loop-invariant Q A-fragments for head chunks k0 = 0, 32.
    v16h aq[2];
    #pragma unroll
    for (int ks = 0; ks < 2; ++ks) {
        const _Float16* p = qrow + (size_t)(qbase + n16) * HEAD + ks * 32 + half * 16;
        v8h lo = *(const v8h*)p;
        v8h hi = *(const v8h*)(p + 8);
        #pragma unroll
        for (int i = 0; i < 8; ++i) { aq[ks][i] = lo[i]; aq[ks][i + 8] = hi[i]; }
    }

    float m[8], l[8];
    v8f oacc[4] = {};
    #pragma unroll
    for (int i = 0; i < 8; ++i) { m[i] = -3.0e38f; l[i] = 0.0f; }

    _Float16* P = &Pbuf[wid][0];
    const int ntiles = (qbase + 16 + 31) >> 5;         // causal: skip upper tiles

    for (int t = 0; t < ntiles; ++t) {
        const int s0 = t << 5;
        if (t + 1 < ntiles) {                          // hint next K/V tiles
            __builtin_prefetch(krow + (size_t)(s0 + 32) * HEAD, 0, 0);
            __builtin_prefetch(vtb + (size_t)n16 * TSEQ + s0 + 32, 0, 0);
        }

        // ---- S = Q K^T  (two 16-col tiles, K-dim = head = 2 x 32) ----
        v8f s[2] = {};
        #pragma unroll
        for (int ks = 0; ks < 2; ++ks) {
            #pragma unroll
            for (int j = 0; j < 2; ++j) {
                const _Float16* kp = krow + (size_t)(s0 + j * 16 + n16) * HEAD
                                          + ks * 32 + half * 16;
                v16h kb = *(const v16h*)kp;
                s[j] = wmma_f16(aq[ks], kb, s[j]);
            }
        }

        // ---- online softmax (row = (vgpr, lane-half); 16 cols across lanes)
        const bool needmask = (s0 + 31) > (qbase + 15);
        float p0[8], p1[8];
        #pragma unroll
        for (int i = 0; i < 8; ++i) {
            float r0 = s[0][i] * 0.125f;               // 1/sqrt(64)
            float r1 = s[1][i] * 0.125f;
            if (needmask) {
                const int q = qbase + i + 8 * half;
                if (s0 + n16 > q)      r0 = -3.0e38f;
                if (s0 + 16 + n16 > q) r1 = -3.0e38f;
            }
            float mx = fmaxf(r0, r1);                  // row-max over 16 lanes
            mx = fmaxf(mx, __shfl_xor(mx, 1));
            mx = fmaxf(mx, __shfl_xor(mx, 2));
            mx = fmaxf(mx, __shfl_xor(mx, 4));
            mx = fmaxf(mx, __shfl_xor(mx, 8));
            const float mnew  = fmaxf(m[i], mx);
            const float alpha = __expf(m[i] - mnew);
            m[i] = mnew;
            const float e0 = __expf(r0 - mnew);
            const float e1 = __expf(r1 - mnew);
            p0[i] = e0; p1[i] = e1;
            float rs = e0 + e1;                        // row-sum over 16 lanes
            rs += __shfl_xor(rs, 1);
            rs += __shfl_xor(rs, 2);
            rs += __shfl_xor(rs, 4);
            rs += __shfl_xor(rs, 8);
            l[i] = l[i] * alpha + rs;
            #pragma unroll
            for (int nt = 0; nt < 4; ++nt) oacc[nt][i] *= alpha;
        }

        // ---- D-layout -> A-layout via per-wave LDS bounce ----
        #pragma unroll
        for (int i = 0; i < 8; ++i) {
            P[(i + 8 * half) * 32 + n16]      = (_Float16)p0[i];
            P[(i + 8 * half) * 32 + 16 + n16] = (_Float16)p1[i];
        }
        asm volatile("s_wait_dscnt 0x0" ::: "memory");
        const _Float16* pp = P + n16 * 32 + half * 16;
        v8h lo = *(const v8h*)pp;
        v8h hi = *(const v8h*)(pp + 8);
        v16h ap;
        #pragma unroll
        for (int i = 0; i < 8; ++i) { ap[i] = lo[i]; ap[i + 8] = hi[i]; }

        // ---- O += P V  (contiguous B-frags thanks to transposed V) ----
        #pragma unroll
        for (int nt = 0; nt < 4; ++nt) {
            const _Float16* vp = vtb + (size_t)(nt * 16 + n16) * TSEQ
                                     + s0 + half * 16;
            v16h vb = *(const v16h*)vp;
            oacc[nt] = wmma_f16(ap, vb, oacc[nt]);
        }
    }

    // ---- normalize and write out (f32, row-major) ----
    float* orow = out + ((size_t)b * TSEQ + qbase) * HEAD;
    #pragma unroll
    for (int nt = 0; nt < 4; ++nt)
        #pragma unroll
        for (int i = 0; i < 8; ++i)
            orow[(size_t)(i + 8 * half) * HEAD + nt * 16 + n16] =
                oacc[nt][i] * (1.0f / l[i]);
}

// ------------------------------------------------------------------- launch
extern "C" void kernel_launch(void* const* d_in, const int* in_sizes, int n_in,
                              void* d_out, int out_size, void* d_ws, size_t ws_size,
                              hipStream_t stream) {
    const float* x  = (const float*)d_in[0];
    const float* Wq = (const float*)d_in[1];
    const float* Wk = (const float*)d_in[2];
    const float* Wv = (const float*)d_in[3];

    char* ws = (char*)d_ws;
    _Float16* wt = (_Float16*)(ws);                                   // 384 KB
    _Float16* qf = (_Float16*)(ws + (size_t)(512 << 10));             // 2 MB
    _Float16* kf = (_Float16*)(ws + (size_t)(512 << 10) + (2u << 20));// 2 MB
    _Float16* vt = (_Float16*)(ws + (size_t)(512 << 10) + (4u << 20));// 2 MB

    attn_wconv_kernel<<<(192 * 1024) / 256, 256, 0, stream>>>(Wq, Wk, Wv, wt);
    attn_qkv_kernel<<<(BATCH * TSEQ) / 16, 96, 0, stream>>>(x, wt, qf, kf, vt);
    attn_flash_kernel<<<(BATCH * TSEQ) / 16 / 4, 128, 0, stream>>>(
        qf, kf, vt, (float*)d_out);
}
